// BiLSTMCRF_28346784154012
// MI455X (gfx1250) — compile-verified, matrix operand order checked by reading, not run on hardware
//
#include <hip/hip_runtime.h>
#include <math.h>

// ---------------------------------------------------------------------------
// BiLSTM-CRF forward for MI455X (gfx1250, wave32, WMMA bf16 16x16x32).
// L=256, B=64, EMB=300, CEMB=32, NF=4, HID=512, NTAGS=32.
// ---------------------------------------------------------------------------

#define SENT 256
#define BATCH 64
#define EMB 300
#define CEMB 32
#define NFILT 128          // CEMB*NF
#define WLEN 16
#define HID 512
#define GATES 2048         // 4*HID
#define NTAGS 32
#define TAG_PAD 31
#define ROWS (SENT*BATCH)  // 16384
#define K0PAD 448          // 428 padded to mult of 32
#define HLDS_STRIDE 528    // 512 + 16: 32B-aligned rows, staggered banks
#define HBUF (BATCH*HLDS_STRIDE)

typedef __attribute__((ext_vector_type(16))) __bf16 v16bf;
typedef __attribute__((ext_vector_type(8)))  float  v8f;

static __device__ __forceinline__ unsigned short f2bf(float f) {
  union { float f; unsigned int u; } v; v.f = f;
  unsigned int u = v.u;
  u += 0x7FFFu + ((u >> 16) & 1u);   // round-to-nearest-even
  return (unsigned short)(u >> 16);
}
static __device__ __forceinline__ float sigmoidf_(float x) {
  return 1.0f / (1.0f + __expf(-x));
}

// --------------------------- small utility kernels -------------------------

__global__ void k_zero_loss(float* out) { out[ROWS] = 0.0f; }

// fp32 -> bf16 with optional column padding (rows = gridDim.x)
__global__ void k_cvt_pad(const float* __restrict__ src, unsigned short* __restrict__ dst,
                          int scols, int dcols) {
  int r = blockIdx.x;
  for (int c = threadIdx.x; c < dcols; c += blockDim.x)
    dst[(size_t)r * dcols + c] = (c < scols) ? f2bf(src[(size_t)r * scols + c]) : 0;
}

__global__ void k_bias_add(const float* __restrict__ a, const float* __restrict__ b,
                           float* __restrict__ dst, int n) {
  int i = blockIdx.x * blockDim.x + threadIdx.x;
  if (i < n) dst[i] = a[i] + b[i];
}

// ------------------- embedding gather + grouped char CNN -------------------
// One block per token (row r = l*64+b). Produces bf16 X [ROWS, 448]:
// cols 0..299 word emb, 300..427 char-CNN maxpool, 428..447 zero pad.
__global__ void k_build_input(const int* __restrict__ words, const float* __restrict__ wemb,
                              const int* __restrict__ chars, const float* __restrict__ cemb,
                              const float* __restrict__ cw, const float* __restrict__ cb,
                              unsigned short* __restrict__ X) {
  int r = blockIdx.x;
  int l = r >> 6, b = r & 63;
  __shared__ float ce[WLEN][CEMB];
  for (int i = threadIdx.x; i < WLEN * CEMB; i += blockDim.x) {
    int w = i >> 5, c = i & 31;
    int ch = chars[((size_t)b * SENT + l) * WLEN + w];
    ce[w][c] = cemb[ch * CEMB + c];
  }
  int wid = words[l * BATCH + b];
  unsigned short* xr = X + (size_t)r * K0PAD;
  for (int i = threadIdx.x; i < EMB; i += blockDim.x)
    xr[i] = f2bf(wemb[(size_t)wid * EMB + i]);
  for (int i = threadIdx.x; i < K0PAD - (EMB + NFILT); i += blockDim.x)
    xr[EMB + NFILT + i] = 0;
  __syncthreads();
  // grouped conv1d (groups=32, 4 out ch per group, k=3, VALID) + max over time
  int o = threadIdx.x;                 // blockDim.x == 128 == NFILT
  int ic = o >> 2;
  float w0 = cw[o * 3], w1 = cw[o * 3 + 1], w2 = cw[o * 3 + 2], bb = cb[o];
  float mx = -1e30f;
  for (int t = 0; t < WLEN - 2; t++) {
    float v = w0 * ce[t][ic] + w1 * ce[t + 1][ic] + w2 * ce[t + 2][ic] + bb;
    mx = fmaxf(mx, v);
  }
  xr[EMB + o] = f2bf(mx);
}

// ------------------------- WMMA bf16 tile GEMM -----------------------------
// C[M,N] = A[M,K](bf16) * B[N,K](bf16)^T + bias[N].
// Each wave owns a 16x64 output strip (4 N-tiles sharing one A-tile load ->
// ~1.25 b128 loads per WMMA). grid.x = M/16; wave strip = blockIdx.y*8+wid.
// NTSTORE: stream C past L2 (G buffers are 134MB write-once/read-once; keep
// the 2MB Whh weights resident instead).
template <bool NTSTORE>
__global__ void k_gemm(const unsigned short* __restrict__ A, int lda,
                       const unsigned short* __restrict__ B, int ldb,
                       const float* __restrict__ bias, float* __restrict__ C, int ldc,
                       int ktiles, int ntiles) {
  int wid = threadIdx.x >> 5, lane = threadIdx.x & 31;
  int mt = blockIdx.x;
  int nt0 = (blockIdx.y * 8 + wid) * 4;
  if (nt0 >= ntiles) return;                      // wave-uniform exit
  int nact = ntiles - nt0; if (nact > 4) nact = 4; // wave-uniform
  int l15 = lane & 15, lhi = lane >> 4;
  const unsigned short* ap = A + (size_t)(mt * 16 + l15) * lda + lhi * 16;
  const unsigned short* bp = B + (size_t)(nt0 * 16 + l15) * ldb + lhi * 16;
  size_t bstride = (size_t)16 * ldb;              // one N-tile of B rows
  v8f acc[4] = {v8f{}, v8f{}, v8f{}, v8f{}};
  for (int k = 0; k < ktiles; k++) {
    v16bf a = *(const v16bf*)(ap); ap += 32;
    #pragma unroll
    for (int j = 0; j < 4; j++) {
      if (j < nact) {
        v16bf bm = *(const v16bf*)(bp + (size_t)j * bstride);
        acc[j] = __builtin_amdgcn_wmma_f32_16x16x32_bf16(false, a, false, bm,
                                                         (short)0, acc[j], false, false);
      }
    }
    bp += 32;
  }
  int rb = mt * 16 + lhi * 8;
  #pragma unroll
  for (int j = 0; j < 4; j++) {
    if (j < nact) {
      int col = (nt0 + j) * 16 + l15;
      float bv = bias[col];
      #pragma unroll
      for (int v = 0; v < 8; v++) {
        float val = acc[j][v] + bv;
        float* cp = &C[(size_t)(rb + v) * ldc + col];
        if (NTSTORE) __builtin_nontemporal_store(val, cp);
        else         *cp = val;
      }
    }
  }
}

// --------------------------- persistent LSTM scan --------------------------
// grid.x = 2 (dir 0=fwd, 1=bwd), block = 512 threads = 16 waves
// (4 waves/SIMD -> ~256 VGPR budget, no spills).
// G[dir]: precomputed x@Wih^T + b, fp32 [L,64,2048] (i|f|g|o blocks of 512).
// Whh bf16 [2048,512]: 2MB, L2-resident, re-streamed from cache EVERY step.
// A 32-bit zero offset is laundered through inline-asm each timestep and
// added to W: the loads stay non-loop-invariant (no LICM->scratch spills)
// while W keeps its GLOBAL address space (global_load_b128, independent
// LOADcnt vs the LDS DScnt stream; flat would couple both counters).
// h is DOUBLE-BUFFERED bf16 in LDS (read h(t-1) from buf cur, write h(t) to
// buf cur^1 immediately -> one barrier per step). c lives in VGPRs in the
// native WMMA D-layout. Each wave computes matching i/f/g/o tiles for its 16
// cell columns: 4 WMMAs per A-tile load, cell update entirely in registers.
// H out bf16 [L*64, 1024], fwd cols 0..511, bwd 512..1023.
__global__ void __launch_bounds__(512, 1)
k_scan(const float* __restrict__ Gf, const float* __restrict__ Gb,
       const unsigned short* __restrict__ Whf, const unsigned short* __restrict__ Whb,
       unsigned short* __restrict__ H) {
  int dir = blockIdx.x;
  const float* G = dir ? Gb : Gf;
  const unsigned short* W = dir ? Whb : Whf;
  int colofs = dir ? HID : 0;
  __shared__ unsigned short hl[2 * HBUF];       // 132 KB (<320KB/WG on gfx1250)
  int tid = threadIdx.x, wid = tid >> 5, lane = tid & 31;
  int l15 = lane & 15, lhi = lane >> 4;
  for (int i = tid; i < 2 * HBUF; i += 512) hl[i] = 0;
  __syncthreads();
  float c[8][8];
  #pragma unroll
  for (int g = 0; g < 8; g++)
    #pragma unroll
    for (int v = 0; v < 8; v++) c[g][v] = 0.0f;

  int cur = 0;
  for (int s = 0; s < SENT; s++) {
    int t = dir ? (SENT - 1 - s) : s;
    const unsigned short* hr = hl + cur * HBUF;
    unsigned short* hw = hl + (cur ^ 1) * HBUF;
    // Launder a zero offset so the W loads are not loop-invariant (prevents
    // LICM + spill) while preserving W's global address space.
    unsigned wofs = 0;
    asm volatile("" : "+s"(wofs));
    const unsigned short* Wt = W + wofs;
    #pragma unroll
    for (int g = 0; g < 8; g++) {
      int gg = wid * 8 + g;
      int mi = gg >> 5, nj = gg & 31;          // 4 M-tiles x 32 N(cell)-tiles
      v8f ai = {}, af = {}, ag = {}, ao = {};
      const unsigned short* ap = &hr[(mi * 16 + l15) * HLDS_STRIDE + lhi * 16];
      const unsigned short* bp = Wt + (size_t)(nj * 16 + l15) * HID + lhi * 16;
      for (int k = 0; k < 16; k++) {
        v16bf a  = *(const v16bf*)(ap + k * 32);
        v16bf bi = *(const v16bf*)(bp + k * 32);
        v16bf bf = *(const v16bf*)(bp + (size_t)512 * HID + k * 32);
        v16bf bg = *(const v16bf*)(bp + (size_t)1024 * HID + k * 32);
        v16bf bo = *(const v16bf*)(bp + (size_t)1536 * HID + k * 32);
        ai = __builtin_amdgcn_wmma_f32_16x16x32_bf16(false, a, false, bi, (short)0, ai, false, false);
        af = __builtin_amdgcn_wmma_f32_16x16x32_bf16(false, a, false, bf, (short)0, af, false, false);
        ag = __builtin_amdgcn_wmma_f32_16x16x32_bf16(false, a, false, bg, (short)0, ag, false, false);
        ao = __builtin_amdgcn_wmma_f32_16x16x32_bf16(false, a, false, bo, (short)0, ao, false, false);
      }
      int col = nj * 16 + l15;
      const float* gp = G + ((size_t)t * BATCH + mi * 16 + lhi * 8) * GATES;
      #pragma unroll
      for (int v = 0; v < 8; v++) {
        size_t ro = (size_t)v * GATES;
        float gi = ai[v] + __builtin_nontemporal_load(gp + ro + col);
        float gf = af[v] + __builtin_nontemporal_load(gp + ro + 512 + col);
        float gc = ag[v] + __builtin_nontemporal_load(gp + ro + 1024 + col);
        float go = ao[v] + __builtin_nontemporal_load(gp + ro + 1536 + col);
        float cn = sigmoidf_(gf) * c[g][v] + sigmoidf_(gi) * tanhf(gc);
        c[g][v] = cn;
        float hh = sigmoidf_(go) * tanhf(cn);
        unsigned short hb = f2bf(hh);
        int row = mi * 16 + lhi * 8 + v;
        hw[row * HLDS_STRIDE + col] = hb;      // other buffer: no hazard
        H[((size_t)t * BATCH + row) * (2 * HID) + colofs + col] = hb;
      }
    }
    __syncthreads();            // h(t) fully published before next step reads
    cur ^= 1;
  }
}

// -------------------------------- CRF --------------------------------------
// One block per batch element, 32 threads (one per tag).
__global__ void k_crf(const float* __restrict__ em, const int* __restrict__ tags,
                      const float* __restrict__ start, const float* __restrict__ endw,
                      const float* __restrict__ trans, int* __restrict__ hist,
                      float* __restrict__ out) {
  int b = blockIdx.x, t = threadIdx.x;
  __shared__ float sc[NTAGS], vs[NTAGS];
  sc[t] = start[t] + em[((size_t)0 * BATCH + b) * NTAGS + t];
  vs[t] = sc[t];
  __syncthreads();
  for (int l = 1; l < SENT; l++) {
    float e = em[((size_t)l * BATCH + b) * NTAGS + t];
    int mk = (tags[l * BATCH + b] != TAG_PAD);
    float m = -1e30f;
    for (int p = 0; p < NTAGS; p++) m = fmaxf(m, sc[p] + trans[p * NTAGS + t]);
    float s = 0.0f;
    for (int p = 0; p < NTAGS; p++) s += __expf(sc[p] + trans[p * NTAGS + t] - m);
    float nsc = m + __logf(s) + e;
    float best = -1e30f; int barg = 0;
    for (int p = 0; p < NTAGS; p++) {
      float v = vs[p] + trans[p * NTAGS + t];
      if (v > best) { best = v; barg = p; }
    }
    float nvs = best + e;
    hist[((size_t)(l - 1) * BATCH + b) * NTAGS + t] = barg;
    __syncthreads();
    if (mk) { sc[t] = nsc; vs[t] = nvs; }
    __syncthreads();
  }
  if (t == 0) {
    // normalizer
    float m = -1e30f;
    for (int p = 0; p < NTAGS; p++) m = fmaxf(m, sc[p] + endw[p]);
    float s = 0.0f;
    for (int p = 0; p < NTAGS; p++) s += __expf(sc[p] + endw[p] - m);
    float nz = m + __logf(s);
    // viterbi terminal
    float best = -1e30f; int last = 0;
    for (int p = 0; p < NTAGS; p++) {
      float v = vs[p] + endw[p];
      if (v > best) { best = v; last = p; }
    }
    // gold path score
    int t0 = tags[b];
    float psc = start[t0] + em[(size_t)b * NTAGS + t0];
    int prev = t0;
    for (int l = 1; l < SENT; l++) {
      int tg = tags[l * BATCH + b];
      float mk = (tg != TAG_PAD) ? 1.0f : 0.0f;
      psc += (trans[prev * NTAGS + tg] + em[((size_t)l * BATCH + b) * NTAGS + tg]) * mk;
      if (mk > 0.0f) prev = tg;
    }
    psc += endw[prev];
    atomicAdd(out + ROWS, nz - psc);            // loss = sum(normalizer - score)
    // backtrack
    int cur = last;
    for (int j = SENT - 2; j >= 0; j--) {
      int l = j + 1;
      int mk = (tags[l * BATCH + b] != TAG_PAD);
      out[l * BATCH + b] = mk ? (float)cur : 0.0f;
      if (mk) cur = hist[((size_t)j * BATCH + b) * NTAGS + cur];
    }
    out[b] = (tags[b] != TAG_PAD) ? (float)cur : 0.0f;
  }
}

// ------------------------------ host launcher ------------------------------

extern "C" void kernel_launch(void* const* d_in, const int* in_sizes, int n_in,
                              void* d_out, int out_size, void* d_ws, size_t ws_size,
                              hipStream_t stream) {
  const int*   words = (const int*)d_in[0];
  const int*   chars = (const int*)d_in[1];
  const int*   tags  = (const int*)d_in[2];
  const float* wemb  = (const float*)d_in[3];
  const float* cemb  = (const float*)d_in[4];
  const float* cnnw  = (const float*)d_in[5];
  const float* cnnb  = (const float*)d_in[6];
  // params order per layer: Wih, Whh, bih, bhh for d in (f,b)
  const float* Wih[2][2]; const float* Whh[2][2];
  const float* bih[2][2]; const float* bhh[2][2];
  int idx = 7;
  for (int l = 0; l < 2; l++)
    for (int d = 0; d < 2; d++) {
      Wih[l][d] = (const float*)d_in[idx++];
      Whh[l][d] = (const float*)d_in[idx++];
      bih[l][d] = (const float*)d_in[idx++];
      bhh[l][d] = (const float*)d_in[idx++];
    }
  const float* fcw   = (const float*)d_in[idx++];
  const float* fcb   = (const float*)d_in[idx++];
  const float* cstart= (const float*)d_in[idx++];
  const float* cend  = (const float*)d_in[idx++];
  const float* ctrans= (const float*)d_in[idx++];
  float* out = (float*)d_out;

  // workspace carve-up (256B aligned)
  char* p = (char*)d_ws;
  size_t off = 0;
  auto alloc = [&](size_t bytes) {
    void* r = p + off;
    off += (bytes + 255) & ~(size_t)255;
    return r;
  };
  unsigned short* Xh  = (unsigned short*)alloc((size_t)ROWS * K0PAD * 2);
  float* G0f          = (float*)alloc((size_t)ROWS * GATES * 4);
  float* G0b          = (float*)alloc((size_t)ROWS * GATES * 4);
  unsigned short* H0  = (unsigned short*)alloc((size_t)ROWS * 2 * HID * 2);
  unsigned short* H1  = (unsigned short*)alloc((size_t)ROWS * 2 * HID * 2);
  float* em           = (float*)alloc((size_t)ROWS * NTAGS * 4);
  unsigned short* Wih_h[2][2]; unsigned short* Whh_h[2][2]; float* bias_c[2][2];
  for (int l = 0; l < 2; l++)
    for (int d = 0; d < 2; d++) {
      Wih_h[l][d] = (unsigned short*)alloc((size_t)GATES * (l ? 1024 : K0PAD) * 2);
      Whh_h[l][d] = (unsigned short*)alloc((size_t)GATES * HID * 2);
      bias_c[l][d] = (float*)alloc((size_t)GATES * 4);
    }
  unsigned short* fcw_h = (unsigned short*)alloc((size_t)NTAGS * 1024 * 2);
  int* hist             = (int*)alloc((size_t)(SENT - 1) * BATCH * NTAGS * 4);
  (void)ws_size; (void)n_in; (void)in_sizes; (void)out_size;

  k_zero_loss<<<1, 1, 0, stream>>>(out);

  // weight conversions (fp32 -> bf16, K-padded)
  for (int l = 0; l < 2; l++)
    for (int d = 0; d < 2; d++) {
      int sk = l ? 1024 : 428, dk = l ? 1024 : K0PAD;
      k_cvt_pad<<<GATES, 256, 0, stream>>>(Wih[l][d], Wih_h[l][d], sk, dk);
      k_cvt_pad<<<GATES, 256, 0, stream>>>(Whh[l][d], Whh_h[l][d], HID, HID);
      k_bias_add<<<GATES / 256, 256, 0, stream>>>(bih[l][d], bhh[l][d], bias_c[l][d], GATES);
    }
  k_cvt_pad<<<NTAGS, 256, 0, stream>>>(fcw, fcw_h, 1024, 1024);

  // embeddings + char CNN -> bf16 input [ROWS, 448]
  k_build_input<<<ROWS, NFILT, 0, stream>>>(words, wemb, chars, cemb, cnnw, cnnb, Xh);

  // layer 0: gate pre-projection GEMMs (streamed C), then bidirectional scans
  dim3 gg(ROWS / 16, GATES / 512);   // 8 waves x 4 N-tiles x 16 = 512 cols/block
  k_gemm<true><<<gg, 256, 0, stream>>>(Xh, K0PAD, Wih_h[0][0], K0PAD, bias_c[0][0], G0f, GATES, K0PAD / 32, GATES / 16);
  k_gemm<true><<<gg, 256, 0, stream>>>(Xh, K0PAD, Wih_h[0][1], K0PAD, bias_c[0][1], G0b, GATES, K0PAD / 32, GATES / 16);
  k_scan<<<2, 512, 0, stream>>>(G0f, G0b, Whh_h[0][0], Whh_h[0][1], H0);

  // layer 1 (reuse G buffers)
  k_gemm<true><<<gg, 256, 0, stream>>>(H0, 1024, Wih_h[1][0], 1024, bias_c[1][0], G0f, GATES, 1024 / 32, GATES / 16);
  k_gemm<true><<<gg, 256, 0, stream>>>(H0, 1024, Wih_h[1][1], 1024, bias_c[1][1], G0b, GATES, 1024 / 32, GATES / 16);
  k_scan<<<2, 512, 0, stream>>>(G0f, G0b, Whh_h[1][0], Whh_h[1][1], H1);

  // emission scores: [ROWS,1024] x [32,1024]^T + fc_b (em is reused -> cached)
  dim3 ge(ROWS / 16, 1);
  k_gemm<false><<<ge, 256, 0, stream>>>(H1, 1024, fcw_h, 1024, fcb, em, NTAGS, 1024 / 32, NTAGS / 16);

  // CRF: loss + viterbi decode
  k_crf<<<BATCH, NTAGS, 0, stream>>>(em, tags, cstart, cend, ctrans, hist, out);
}
